// MockMixtralMoELayer_87995289960529
// MI455X (gfx1250) — matrix operand order repaired and checked by reading, not compiled, not run on hardware
//
#include <hip/hip_runtime.h>
#include <hip/hip_bf16.h>

#define H   1024
#define MT  16      // tokens (M rows) per block
#define NE  8       // num experts

typedef __attribute__((ext_vector_type(16))) __bf16 v16bf;
typedef __attribute__((ext_vector_type(8)))  float  v8f;

union Frag32B { uint4 q[2]; v16bf v; };

__device__ __forceinline__ unsigned short f32_to_bf16_rne(float f) {
  unsigned int u = __float_as_uint(f);
  u += 0x7FFFu + ((u >> 16) & 1u);
  return (unsigned short)(u >> 16);
}

// ---------------------------------------------------------------------------
// Kernel 0: pack expert_weight [K=1024][N=1024] fp32 into WMMA-B-fragment
// order, bf16.  Flat index: (((nt*32 + kb)*32 + lane)*16 + j)
//   N = nt*16 + (lane&15),  K = kb*32 + 16*(lane>>4) + j
// so in the GEMM each lane's B fragment is one contiguous 32-byte load.
// ---------------------------------------------------------------------------
__global__ __launch_bounds__(256) void pack_weight_bf16(
    const float* __restrict__ W, unsigned short* __restrict__ Wtp) {
  unsigned g    = blockIdx.x * 256u + threadIdx.x;   // 0 .. 1M-1
  unsigned j    = g & 15u;
  unsigned lane = (g >> 4) & 31u;
  unsigned kb   = (g >> 9) & 31u;
  unsigned nt   = g >> 14;
  unsigned n    = nt * 16u + (lane & 15u);
  unsigned k    = kb * 32u + 16u * (lane >> 4) + j;
  Wtp[g] = f32_to_bf16_rne(W[k * H + n]);
}

// ---------------------------------------------------------------------------
// Kernel 1: fused gating + shared-expert GEMM (bf16 WMMA) + scale + LayerNorm.
// Block = 256 threads (8 waves). Handles 16 tokens x all 1024 output cols.
// Wave w owns N-columns [128w, 128w+128) as 8 WMMA tiles.
// ---------------------------------------------------------------------------
__global__ __launch_bounds__(256, 1) void moe_fused_wmma(
    const float* __restrict__ X, const float* __restrict__ gateW,
    const unsigned short* __restrict__ Wtp,
    const float* __restrict__ gamma, const float* __restrict__ beta,
    float* __restrict__ Out) {
  __shared__ __align__(16) unsigned short Ash[MT][H + 8]; // bf16 A tile (+pad)
  __shared__ float gpart[256][NE];      // gating partials
  __shared__ float logitsSh[MT][NE];
  __shared__ float scaleSh[MT];
  __shared__ float redSum[8][MT];
  __shared__ float redSq[8][MT];
  __shared__ float muSh[MT];
  __shared__ float rsSh[MT];

  const int tid  = threadIdx.x;
  const int lane = tid & 31;
  const int wave = tid >> 5;
  const int half = lane >> 4;   // 0/1
  const int lrow = lane & 15;   // row (A) / col (B) within tile
  const long tokBase = (long)blockIdx.x * MT;

  // ---- Stage 1: X tile -> bf16 LDS, and gate logits (fused, once/token) ----
  {
    const int row = tid >> 4;            // 0..15 token row
    const int c0  = (tid & 15) * 64;     // 64 floats per thread
    const float* xp = X + (tokBase + row) * H + c0;
    float acc[NE];
#pragma unroll
    for (int e = 0; e < NE; ++e) acc[e] = 0.f;
#pragma unroll
    for (int i = 0; i < 16; ++i) {       // 16 x float4
      float4 v = reinterpret_cast<const float4*>(xp)[i];
      const int c = c0 + i * 4;
      unsigned p0 = (unsigned)f32_to_bf16_rne(v.x) |
                    ((unsigned)f32_to_bf16_rne(v.y) << 16);
      unsigned p1 = (unsigned)f32_to_bf16_rne(v.z) |
                    ((unsigned)f32_to_bf16_rne(v.w) << 16);
      *reinterpret_cast<uint2*>(&Ash[row][c]) = make_uint2(p0, p1);
#pragma unroll
      for (int e = 0; e < NE; ++e) {
        float4 g = reinterpret_cast<const float4*>(gateW + e * H + c)[0];
        acc[e] += v.x * g.x + v.y * g.y + v.z * g.z + v.w * g.w;
      }
    }
#pragma unroll
    for (int e = 0; e < NE; ++e) gpart[tid][e] = acc[e];
  }
  __syncthreads();
  if (tid < MT * NE) {                   // 128 threads: reduce logits
    const int row = tid >> 3, e = tid & 7;
    float s = 0.f;
#pragma unroll
    for (int i = 0; i < 16; ++i) s += gpart[row * 16 + i][e];
    logitsSh[row][e] = s;
  }
  __syncthreads();
  if (tid < MT) {                        // top-2 sum of raw logits
    float m1 = -3.4e38f, m2 = -3.4e38f;
#pragma unroll
    for (int e = 0; e < NE; ++e) {
      float v = logitsSh[tid][e];
      if (v > m1) { m2 = m1; m1 = v; } else if (v > m2) { m2 = v; }
    }
    scaleSh[tid] = m1 + m2;
  }
  __syncthreads();

  // ---- Stage 2: bf16 WMMA GEMM, K = 1024 in steps of 32 ----
  v8f acc[8];
#pragma unroll
  for (int nt = 0; nt < 8; ++nt)
#pragma unroll
    for (int i = 0; i < 8; ++i) acc[nt][i] = 0.f;

  const unsigned short* arow = &Ash[lrow][0];
  for (int kb = 0; kb < H / 32; ++kb) {
    Frag32B a;
    // A layout (matches packed B): elems 0..15 -> K = kb*32+16*half+{0..15}
    const uint4* ap = reinterpret_cast<const uint4*>(arow + kb * 32 + 16 * half);
    a.q[0] = ap[0];
    a.q[1] = ap[1];
#pragma unroll
    for (int nt = 0; nt < 8; ++nt) {
      Frag32B b;
      const uint4* bp = reinterpret_cast<const uint4*>(
          Wtp + ((((unsigned)(wave * 8 + nt)) * 32u + (unsigned)kb) * 32u +
                 (unsigned)lane) * 16u);
      b.q[0] = bp[0];
      b.q[1] = bp[1];
      acc[nt] = __builtin_amdgcn_wmma_f32_16x16x32_bf16(
          false, a.v, false, b.v, (short)0, acc[nt], false, false);
    }
  }

  // ---- Stage 3: routing scale + fused LayerNorm ----
  float srow[8], psum[8], psq[8];
#pragma unroll
  for (int v = 0; v < 8; ++v) {
    srow[v] = scaleSh[v + 8 * half];     // C layout: VGPR v -> row v+8*half
    psum[v] = 0.f;
    psq[v]  = 0.f;
  }
#pragma unroll
  for (int nt = 0; nt < 8; ++nt)
#pragma unroll
    for (int v = 0; v < 8; ++v) {
      float x = acc[nt][v] * srow[v];
      acc[nt][v] = x;
      psum[v] += x;
      psq[v]  += x * x;
    }
  // reduce across the 16 lanes that share a half (same rows)
#pragma unroll
  for (int off = 1; off < 16; off <<= 1)
#pragma unroll
    for (int v = 0; v < 8; ++v) {
      psum[v] += __shfl_xor(psum[v], off, 32);
      psq[v]  += __shfl_xor(psq[v],  off, 32);
    }
  if (lrow == 0) {
#pragma unroll
    for (int v = 0; v < 8; ++v) {
      redSum[wave][v + 8 * half] = psum[v];
      redSq [wave][v + 8 * half] = psq[v];
    }
  }
  __syncthreads();
  if (tid < MT) {
    float ts = 0.f, tq = 0.f;
#pragma unroll
    for (int w = 0; w < 8; ++w) { ts += redSum[w][tid]; tq += redSq[w][tid]; }
    float mu  = ts * (1.0f / H);
    float var = tq * (1.0f / H) - mu * mu;
    muSh[tid] = mu;
    rsSh[tid] = rsqrtf(var + 1e-5f);
  }
  __syncthreads();

#pragma unroll
  for (int nt = 0; nt < 8; ++nt) {
    const int col = wave * 128 + nt * 16 + lrow;
    const float g  = gamma[col];
    const float bt = beta[col];
#pragma unroll
    for (int v = 0; v < 8; ++v) {
      const int row = v + 8 * half;
      Out[(tokBase + row) * H + col] =
          (acc[nt][v] - muSh[row]) * rsSh[row] * g + bt;
    }
  }
}

// ---------------------------------------------------------------------------
extern "C" void kernel_launch(void* const* d_in, const int* in_sizes, int n_in,
                              void* d_out, int out_size, void* d_ws, size_t ws_size,
                              hipStream_t stream) {
  const float* X     = (const float*)d_in[0];  // [4,8192,1024]
  const float* gateW = (const float*)d_in[1];  // [8,1024]
  const float* W     = (const float*)d_in[2];  // [1024,1024]
  const float* gamma = (const float*)d_in[3];  // [1024]
  const float* beta  = (const float*)d_in[4];  // [1024]
  float* Out = (float*)d_out;

  unsigned short* Wtp = (unsigned short*)d_ws;  // 2 MB packed bf16 weights

  pack_weight_bf16<<<(H * H) / 256, 256, 0, stream>>>(W, Wtp);

  const int nTok = in_sizes[0] / H;             // 32768
  moe_fused_wmma<<<nTok / MT, 256, 0, stream>>>(X, gateW, Wtp, gamma, beta, Out);
}
